// SelfCorrelationComputation_81801947119905
// MI455X (gfx1250) — compile-verified, matrix operand order checked by reading, not run on hardware
//
#include <hip/hip_runtime.h>
#include <hip/hip_bf16.h>
#include <math.h>

typedef __attribute__((ext_vector_type(16))) _Float16     v16h;
typedef __attribute__((ext_vector_type(8)))  float        v8f;
typedef __attribute__((ext_vector_type(4)))  float        f4;
typedef __attribute__((ext_vector_type(4)))  unsigned int u32x4;
typedef __attribute__((ext_vector_type(8)))  int          i32x8;
typedef __attribute__((ext_vector_type(4)))  int          i32x4;

#define NPIX  32768          // B*H*W = 8*64*64
#define CCH   96
#define HWSZ  4096
#define HH    64
#define WW2   64

// ---------------- workspace layout (bytes) ----------------
#define XH_OFF   0            // xh   f32 [N,96]   (12,582,912)
#define LN_OFF   12582912     // ln   f16 [N,96]   ( 6,291,456)  LN1 then LN2 output
#define AT_OFF   18874368     // attn f16 [N,96]   ( 6,291,456)
#define QKV_OFF  25165824     // qkv  f32 [N,288]  (37,748,736)  reused: hidden f16, then xcn f32
#define W16_OFF  62914560     // packed f16 weights (221,184)
// f16-element offsets inside the packed weight block
#define WQKV16   0
#define WPROJ16  27648
#define W1_16    36864
#define W2_16    73728
#define WTOT16   110592

#define HAS_TDM __has_builtin(__builtin_amdgcn_tensor_load_to_lds)

// ---------------- weight pack f32 -> f16 ----------------
__global__ void k_prep_weights(const float* Wqkv, const float* Wproj,
                               const float* W1, const float* W2, _Float16* out) {
  int i = blockIdx.x * 256 + threadIdx.x;
  if (i >= WTOT16) return;
  float v;
  if (i < WPROJ16)      v = Wqkv[i];
  else if (i < W1_16)   v = Wproj[i - WPROJ16];
  else if (i < W2_16)   v = W1[i - W1_16];
  else                  v = W2[i - W2_16];
  out[i] = (_Float16)v;
}

// ---------------- transpose + LayerNorm1 ----------------
__global__ void k_ln1(const float* __restrict__ x, const float* __restrict__ g,
                      const float* __restrict__ b, float* __restrict__ xh,
                      _Float16* __restrict__ ln) {
  int n = blockIdx.x * blockDim.x + threadIdx.x;
  if (n >= NPIX) return;
  int bi = n >> 12, hw = n & (HWSZ - 1);
  const float* xp = x + (size_t)bi * CCH * HWSZ + hw;
  float s = 0.f, ss = 0.f;
  for (int c = 0; c < CCH; ++c) { float v = xp[(size_t)c * HWSZ]; s += v; ss += v * v; }
  float mean = s * (1.f / 96.f);
  float var  = ss * (1.f / 96.f) - mean * mean;
  float ri   = rsqrtf(var + 1e-5f);
  float*    xhp = xh + (size_t)n * CCH;
  _Float16* lnp = ln + (size_t)n * CCH;
  for (int c = 0; c < CCH; ++c) {
    float v = xp[(size_t)c * HWSZ];
    xhp[c] = v;
    lnp[c] = (_Float16)((v - mean) * ri * g[c] + b[c]);
  }
}

// ---------------- LayerNorm2 (row-major input) ----------------
__global__ void k_ln2(const float* __restrict__ xh, const float* __restrict__ g,
                      const float* __restrict__ b, _Float16* __restrict__ ln) {
  int n = blockIdx.x * blockDim.x + threadIdx.x;
  if (n >= NPIX) return;
  const float* p = xh + (size_t)n * CCH;
  float s = 0.f, ss = 0.f;
  for (int c = 0; c < CCH; ++c) { float v = p[c]; s += v; ss += v * v; }
  float mean = s * (1.f / 96.f);
  float var  = ss * (1.f / 96.f) - mean * mean;
  float ri   = rsqrtf(var + 1e-5f);
  _Float16* lnp = ln + (size_t)n * CCH;
  for (int c = 0; c < CCH; ++c) lnp[c] = (_Float16)((p[c] - mean) * ri * g[c] + b[c]);
}

// ---------------- WMMA GEMM: [N,K]f16 x [Nout,K]f16^T -> epilogue ----------------
// Block = 256 threads = 8 waves sharing one n-tile. The 16xK weight slab is
// DMA'd into LDS by the Tensor Data Mover (wave 0 issues tensor_load_to_lds,
// waits on TENSORcnt), with TDM pad_enable inserting 16B of padding per 64B so
// the padded row stride (K*2*1.25 B) spreads fragment rows across LDS banks.
// LDS half-offset of stored half kh:  kh + 8*(kh>>5)   (16B-aligned for kh%16==0)
enum { EP_F32 = 0, EP_F16 = 1, EP_GELU_F16 = 2, EP_RES_F32 = 3 };

template <int MODE>
__global__ void k_gemm(const _Float16* __restrict__ A, const _Float16* __restrict__ Bw,
                       const float* __restrict__ bias, float* __restrict__ outF,
                       _Float16* __restrict__ outH, float* __restrict__ resid,
                       int Kdim, int Nout, int ntiles) {
  __shared__ __align__(16) _Float16 sB[16 * 480];   // 16 rows * (384*1.25) halves = 15 KB

  int tn = blockIdx.x % ntiles;
  int bg = blockIdx.x / ntiles;
  int waveId = threadIdx.x >> 5;
  int lane   = threadIdx.x & 31;
  int tm = bg * 8 + waveId;

#if HAS_TDM
  // ---- TDM stage: 2-D tile [16 rows x Kdim halves] -> LDS with padding ----
  if (waveId == 0) {
    unsigned ldsAddr = (unsigned)(size_t)&sB[0];    // flat->LDS truncates to offset
    unsigned long long ga =
        (unsigned long long)(size_t)(Bw + (size_t)tn * 16 * Kdim);
    u32x4 g0;
    g0.x = 1u;                                       // count=1 (valid user descriptor)
    g0.y = ldsAddr;                                  // lds_addr
    g0.z = (unsigned)ga;                             // global_addr[31:0]
    g0.w = ((unsigned)(ga >> 32) & 0x01FFFFFFu)      // global_addr[56:32]
           | (2u << 30);                             // type=2 ("image")
    i32x8 g1;
    g1[0] = (1 << 16)        // data_size = 2 bytes
          | (1 << 20)        // pad_enable
          | (3 << 22)        // pad_interval: every 16 DWORDs (64B)
          | (3 << 25);       // pad_amount: 4 DWORDs (16B)
    g1[1] = (Kdim & 0xFFFF) << 16;                   // tensor_dim0[15:0]
    g1[2] = (Kdim >> 16) | (16 << 16);               // tensor_dim0[31:16] | tensor_dim1[15:0]
    g1[3] = (Kdim & 0xFFFF) << 16;                   // tensor_dim1[31:16]=0 | tile_dim0
    g1[4] = 16;                                      // tile_dim1=16, tile_dim2=0
    g1[5] = Kdim;                                    // tensor_dim0_stride[31:0]
    g1[6] = 0;                                       // stride0[47:32] | stride1[15:0]
    g1[7] = 0;                                       // stride1[47:16]
    i32x4 gz4 = {0, 0, 0, 0};
    i32x8 gz8 = {0, 0, 0, 0, 0, 0, 0, 0};
    __builtin_amdgcn_tensor_load_to_lds(g0, g1, gz4, gz4, gz8, 0);
    __builtin_amdgcn_s_wait_tensorcnt((short)0);
  }
  __syncthreads();
#else
  // ---- manual stage with identical stream-padded layout ----
  {
    const unsigned long long* src =
        reinterpret_cast<const unsigned long long*>(Bw + (size_t)tn * 16 * Kdim);
    unsigned long long* dst = reinterpret_cast<unsigned long long*>(sB);
    int nch = (16 * Kdim) >> 2;                      // 8-byte chunks
    for (int i = threadIdx.x; i < nch; i += 256)
      dst[i + ((i >> 3) << 1)] = src[i];             // +16B pad per 64B stored
  }
  __syncthreads();
#endif

  int rowP = Kdim + (Kdim >> 2);                     // padded halves per LDS row
  int arow = tm * 16 + (lane & 15);
  int ksub = (lane >> 4) << 4;                       // 0 or 16: lane's 16-half K chunk
  const _Float16* bRow = sB + (size_t)(lane & 15) * rowP;
  v8f acc = {};
  for (int k0 = 0; k0 < Kdim; k0 += 32) {
    int kh = k0 + ksub;
    v16h a  = *reinterpret_cast<const v16h*>(A + (size_t)arow * Kdim + kh);
    v16h bb = *reinterpret_cast<const v16h*>(bRow + kh + ((kh >> 5) << 3));
    acc = __builtin_amdgcn_wmma_f32_16x16x32_f16(false, a, false, bb,
                                                 (short)0, acc, false, false);
  }
  int col   = tn * 16 + (lane & 15);
  int rbase = tm * 16 + ((lane >> 4) << 3);
  float bv = bias[col];
#pragma unroll
  for (int r = 0; r < 8; ++r) {
    int row = rbase + r;
    float v = acc[r] + bv;
    if (MODE == EP_F32) {
      outF[(size_t)row * Nout + col] = v;
    } else if (MODE == EP_F16) {
      outH[(size_t)row * Nout + col] = (_Float16)v;
    } else if (MODE == EP_GELU_F16) {
      float gg = 0.5f * v * (1.f + erff(v * 0.70710678118654752f));
      outH[(size_t)row * Nout + col] = (_Float16)gg;
    } else {  // EP_RES_F32: residual add into xh (Nout == 96)
      resid[(size_t)row * CCH + col] += v;
    }
  }
}

// ---------------- dilated 3x3 local attention ----------------
__global__ void k_attn(const float* __restrict__ qkv, _Float16* __restrict__ attnout) {
  int n = blockIdx.x * blockDim.x + threadIdx.x;
  if (n >= NPIX) return;
  int br  = blockIdx.y;          // dilation branch 0..2
  int dil = br + 1;
  int bi = n >> 12, hw = n & (HWSZ - 1), h = hw >> 6, w = hw & 63;
  const float* base = qkv + (size_t)n * 288;
  for (int head = 0; head < 2; ++head) {
    int qc = br * 32 + head * 16;
    float q[16];
#pragma unroll
    for (int d = 0; d < 16; ++d) q[d] = base[qc + d];
    float lg[9];
#pragma unroll
    for (int kk = 0; kk < 9; ++kk) {
      int hh = h + (kk / 3 - 1) * dil, ww = w + (kk % 3 - 1) * dil;
      float dot = 0.f;
      if (hh >= 0 && hh < HH && ww >= 0 && ww < WW2) {
        const float* kp = qkv + ((size_t)(bi * HWSZ + hh * WW2 + ww)) * 288 + 96 + qc;
#pragma unroll
        for (int d = 0; d < 16; ++d) dot += q[d] * kp[d];
      }
      lg[kk] = dot * 0.25f;   // head_dim^-0.5; zero-padded k => logit 0 (kept in softmax)
    }
    float mx = lg[0];
#pragma unroll
    for (int kk = 1; kk < 9; ++kk) mx = fmaxf(mx, lg[kk]);
    float sum = 0.f;
#pragma unroll
    for (int kk = 0; kk < 9; ++kk) { lg[kk] = __expf(lg[kk] - mx); sum += lg[kk]; }
    float rs = 1.f / sum;
    float o[16];
#pragma unroll
    for (int d = 0; d < 16; ++d) o[d] = 0.f;
#pragma unroll
    for (int kk = 0; kk < 9; ++kk) {
      int hh = h + (kk / 3 - 1) * dil, ww = w + (kk % 3 - 1) * dil;
      if (hh >= 0 && hh < HH && ww >= 0 && ww < WW2) {
        const float* vp = qkv + ((size_t)(bi * HWSZ + hh * WW2 + ww)) * 288 + 192 + qc;
        float wgt = lg[kk] * rs;
#pragma unroll
        for (int d = 0; d < 16; ++d) o[d] += wgt * vp[d];
      }
    }
    _Float16* op = attnout + (size_t)n * CCH + qc;
#pragma unroll
    for (int d = 0; d < 16; ++d) op[d] = (_Float16)o[d];
  }
}

// ---------------- relu + channel L2-normalize -> [B,C,H,W] ----------------
__global__ void k_norm(const float* __restrict__ xh, float* __restrict__ xcn) {
  int n = blockIdx.x * blockDim.x + threadIdx.x;
  if (n >= NPIX) return;
  const float* p = xh + (size_t)n * CCH;
  float ss = 0.f;
  for (int c = 0; c < CCH; ++c) { float v = fmaxf(p[c], 0.f); ss += v * v; }
  float r = 1.f / fmaxf(sqrtf(ss), 1e-12f);
  int bi = n >> 12, hw = n & (HWSZ - 1);
  for (int c = 0; c < CCH; ++c)
    xcn[((size_t)bi * CCH + c) * HWSZ + hw] = fmaxf(p[c], 0.f) * r;
}

// ---------------- 5x5 self-correlation ----------------
// Dominant cost: 315 MB output stream. Each wave stages its 32x25 = 800
// contiguous output floats in LDS (25-dword stride => conflict-free), then
// streams them out as coalesced 16B-aligned non-temporal b128 stores.
__global__ void k_selfcorr(const float* __restrict__ xcn, float* __restrict__ out) {
  __shared__ __align__(16) float sOut[8 * 800];   // 25.6 KB
  int tid = threadIdx.x;
  int waveId = tid >> 5, lane = tid & 31;
  size_t t = (size_t)blockIdx.x * 256 + tid;      // over B*C*H*W (exact grid)
  int hw = (int)(t & (HWSZ - 1)); int h = hw >> 6, w = hw & 63;
  size_t bc = t >> 12;                            // b*96 + c
  float self = xcn[t];
  float* sp = &sOut[waveId * 800 + lane * 25];
#pragma unroll
  for (int i = 0; i < 5; ++i) {
    int hhp = h + i - 2;
#pragma unroll
    for (int j = 0; j < 5; ++j) {
      int wwp = w + j - 2;
      float v = 0.f;
      if (hhp >= 0 && hhp < HH && wwp >= 0 && wwp < WW2)
        v = xcn[(bc << 12) + hhp * WW2 + wwp] * self;
      sp[i * 5 + j] = v;
    }
  }
  __syncthreads();
  size_t waveBase = ((size_t)blockIdx.x * 256 + (size_t)(waveId << 5)) * 25;  // 3200B aligned
  const f4* s4 = reinterpret_cast<const f4*>(&sOut[waveId * 800]);
  f4* o4 = reinterpret_cast<f4*>(out + waveBase);
  for (int i = lane; i < 200; i += 32)
    __builtin_nontemporal_store(s4[i], o4 + i);
}

extern "C" void kernel_launch(void* const* d_in, const int* in_sizes, int n_in,
                              void* d_out, int out_size, void* d_ws, size_t ws_size,
                              hipStream_t stream) {
  const float* x     = (const float*)d_in[0];
  const float* g1    = (const float*)d_in[1];
  const float* b1    = (const float*)d_in[2];
  const float* Wqkv  = (const float*)d_in[3];
  const float* bqkv  = (const float*)d_in[4];
  const float* Wproj = (const float*)d_in[5];
  const float* bproj = (const float*)d_in[6];
  const float* g2    = (const float*)d_in[7];
  const float* b2    = (const float*)d_in[8];
  const float* W1    = (const float*)d_in[9];
  const float* bm1   = (const float*)d_in[10];
  const float* W2    = (const float*)d_in[11];
  const float* bm2   = (const float*)d_in[12];
  float* out = (float*)d_out;

  char* ws = (char*)d_ws;
  float*    xh      = (float*)(ws + XH_OFF);
  _Float16* lnb     = (_Float16*)(ws + LN_OFF);
  _Float16* attnout = (_Float16*)(ws + AT_OFF);
  float*    qkv     = (float*)(ws + QKV_OFF);
  _Float16* hidden  = (_Float16*)(ws + QKV_OFF);  // reuse after attention consumed qkv
  float*    xcn     = (float*)(ws + QKV_OFF);     // reuse after MLP2 consumed hidden
  _Float16* w16     = (_Float16*)(ws + W16_OFF);

  // 0. pack weights to f16
  k_prep_weights<<<(WTOT16 + 255) / 256, 256, 0, stream>>>(Wqkv, Wproj, W1, W2, w16);

  // 1. transpose + LN1
  k_ln1<<<NPIX / 256, 256, 0, stream>>>(x, g1, b1, xh, lnb);

  // 2. QKV GEMM: [N,96] x [96,288] -> qkv f32   (256 m-groups x 18 n-tiles)
  {
    int ntiles = 288 / 16;
    k_gemm<EP_F32><<<(NPIX / 128) * ntiles, 256, 0, stream>>>(
        lnb, w16 + WQKV16, bqkv, qkv, nullptr, nullptr, 96, 288, ntiles);
  }

  // 3. dilated attention (3 branches)
  k_attn<<<dim3(NPIX / 256, 3), 256, 0, stream>>>(qkv, attnout);

  // 4. proj GEMM + bias + residual into xh
  {
    int ntiles = 96 / 16;
    k_gemm<EP_RES_F32><<<(NPIX / 128) * ntiles, 256, 0, stream>>>(
        attnout, w16 + WPROJ16, bproj, nullptr, nullptr, xh, 96, 96, ntiles);
  }

  // 5. LN2
  k_ln2<<<NPIX / 256, 256, 0, stream>>>(xh, g2, b2, lnb);

  // 6. MLP1 GEMM + GELU -> hidden f16 [N,384]
  {
    int ntiles = 384 / 16;
    k_gemm<EP_GELU_F16><<<(NPIX / 128) * ntiles, 256, 0, stream>>>(
        lnb, w16 + W1_16, bm1, nullptr, hidden, nullptr, 96, 384, ntiles);
  }

  // 7. MLP2 GEMM + bias + residual into xh (K=384)
  {
    int ntiles = 96 / 16;
    k_gemm<EP_RES_F32><<<(NPIX / 128) * ntiles, 256, 0, stream>>>(
        hidden, w16 + W2_16, bm2, nullptr, nullptr, xh, 384, 96, ntiles);
  }

  // 8. relu + L2-normalize -> xcn [B,C,H,W]
  k_norm<<<NPIX / 256, 256, 0, stream>>>(xh, xcn);

  // 9. self-correlation output (LDS-staged coalesced NT stores)
  {
    size_t tot = (size_t)8 * CCH * HWSZ;
    k_selfcorr<<<(unsigned)(tot / 256), 256, 0, stream>>>(xcn, out);
  }
}
